// S4_Layer_41721312313809
// MI455X (gfx1250) — compile-verified, matrix operand order checked by reading, not run on hardware
//
#include <hip/hip_runtime.h>
#include <hip/hip_bf16.h>
#include <math.h>

// Problem constants (from reference)
#define BB 4
#define LL 2048
#define DM 512
#define DFF 2048
#define NS 16
#define ROWS (BB * LL)   // 8192

typedef _Float16 v16h __attribute__((ext_vector_type(16)));
typedef _Float16 v8h  __attribute__((ext_vector_type(8)));
typedef float    v8f  __attribute__((ext_vector_type(8)));
typedef unsigned int v4u  __attribute__((ext_vector_type(4)));
typedef int          v4i  __attribute__((ext_vector_type(4)));
typedef int          v8it __attribute__((ext_vector_type(8)));

union F16x16 { v16h v; v8h h[2]; };

// ---------------------------------------------------------------------------
// CDNA5 async helpers
// ---------------------------------------------------------------------------
// Per-lane async copy of 16B global -> LDS (tracked by ASYNCcnt).
__device__ __forceinline__ void async_b128(unsigned lds_byte_off, const void* gptr) {
    asm volatile("global_load_async_to_lds_b128 %0, %1, off"
                 :: "v"(lds_byte_off), "v"(gptr)
                 : "memory");
}

__device__ __forceinline__ void wait_asynccnt0() {
    asm volatile("s_wait_asynccnt 0x0" ::: "memory");
}

#if __has_builtin(__builtin_amdgcn_tensor_load_to_lds) && \
    __has_builtin(__builtin_amdgcn_s_wait_tensorcnt)
#define HAVE_TDM 1
// TDM: DMA a 2-D tile (tile_w x tile_h elements, 2-byte elements, row-major,
// row stride = stride_elems) from global memory into LDS at lds_byte_off.
// Descriptor layout per CDNA5 ISA ch.8 (groups 0/1; groups 2/3 zero for 2D).
// This toolchain exposes the 6-arg builtin:
//   (uint32x4 g0, int32x8 g1, int32x4 g2, int32x4 g3, int32x8 g4, i32 cpol)
__device__ __forceinline__ void tdm_load_tile_2d(unsigned lds_byte_off,
                                                 const void* gaddr,
                                                 unsigned tile_w, unsigned tile_h,
                                                 unsigned long long stride_elems,
                                                 unsigned long long td0,
                                                 unsigned long long td1) {
    const unsigned long long ga = (unsigned long long)(uintptr_t)gaddr;
    v4u g0;
    g0.x = 1u;                                            // count=1, user-mode
    g0.y = lds_byte_off;                                  // lds_addr (bytes)
    g0.z = (unsigned)(ga & 0xFFFFFFFFu);                  // global_addr[31:0]
    g0.w = (unsigned)((ga >> 32) & 0x01FFFFFFu)           // global_addr[56:32]
           | 0x80000000u;                                 // type = 2 ("image")
    v8it g1;
    g1[0] = (int)(1u << 16);                              // data_size=1 -> 2B elems
    g1[1] = (int)((td0 & 0xFFFFu) << 16);                 // tensor_dim0[15:0]
    g1[2] = (int)(((td0 >> 16) & 0xFFFFu) |               // tensor_dim0[31:16]
                  ((td1 & 0xFFFFu) << 16));               // tensor_dim1[15:0]
    g1[3] = (int)(((td1 >> 16) & 0xFFFFu) |               // tensor_dim1[31:16]
                  ((tile_w & 0xFFFFu) << 16));            // tile_dim0
    g1[4] = (int)(tile_h & 0xFFFFu);                      // tile_dim1 (tile_dim2=0)
    g1[5] = (int)(stride_elems & 0xFFFFFFFFu);            // dim0_stride[31:0]
    g1[6] = (int)((stride_elems >> 32) & 0xFFFFu);        // dim0_stride[47:32]
    g1[7] = 0;                                            // dim1_stride (unused, 2D)
    v4i  g2 = {0, 0, 0, 0};
    v4i  g3 = {0, 0, 0, 0};
    v8it g4 = {0, 0, 0, 0, 0, 0, 0, 0};
    __builtin_amdgcn_tensor_load_to_lds(g0, g1, g2, g3, g4, 0);
}
#else
#define HAVE_TDM 0
#endif

// ---------------------------------------------------------------------------
// Kernel 1: S4D diagonal SSM via linear recurrence (equivalent to the FFT
// causal convolution in the reference).  One thread per (b, h) channel, 16
// complex states held in registers.  Writes xres = x + s4d(x).
// ---------------------------------------------------------------------------
__global__ void s4d_kernel(const float* __restrict__ x,
                           const float* __restrict__ log_dt,
                           const float* __restrict__ A_re,
                           const float* __restrict__ A_im,
                           const float* __restrict__ C_re,
                           const float* __restrict__ C_im,
                           const float* __restrict__ Dp,
                           float* __restrict__ xres) {
    const int t = blockIdx.x * blockDim.x + threadIdx.x;   // 0 .. B*DM-1
    if (t >= BB * DM) return;
    const int b = t >> 9;       // / DM
    const int h = t & (DM - 1);

    const float dt = __expf(log_dt[h]);
    float lamr[NS], lami[NS], cr[NS], ci[NS], sr[NS], si[NS];
#pragma unroll
    for (int n = 0; n < NS; ++n) {
        const float ar = A_re[h * NS + n];
        const float ai = A_im[h * NS + n];
        const float dr = ar * dt, di = ai * dt;
        const float e  = __expf(dr);
        const float lr = e * __cosf(di);
        const float li = e * __sinf(di);
        // (exp(dtA)-1)/A  =  (lam-1) * conj(A) / |A|^2
        const float nr = lr - 1.0f, ni = li;
        const float inv = 1.0f / (ar * ar + ai * ai);
        const float qr = (nr * ar + ni * ai) * inv;
        const float qi = (ni * ar - nr * ai) * inv;
        const float crr = C_re[h * NS + n];
        const float cii = C_im[h * NS + n];
        cr[n] = 2.0f * (crr * qr - cii * qi);   // fold the 2*Re factor
        ci[n] = 2.0f * (crr * qi + cii * qr);
        lamr[n] = lr; lami[n] = li;
        sr[n] = 0.0f; si[n] = 0.0f;
    }

    const float onePlusD = 1.0f + Dp[h];
    const float* __restrict__ px = x + (size_t)b * LL * DM + h;
    float* __restrict__ po = xres + (size_t)b * LL * DM + h;

    for (int l = 0; l < LL; ++l) {
        const float u = px[(size_t)l * DM];
        float acc = 0.0f;
#pragma unroll
        for (int n = 0; n < NS; ++n) {
            const float nsr = lamr[n] * sr[n] - lami[n] * si[n] + u;
            const float nsi = lamr[n] * si[n] + lami[n] * sr[n];
            sr[n] = nsr; si[n] = nsi;
            acc += cr[n] * nsr - ci[n] * nsi;
        }
        po[(size_t)l * DM] = fmaf(onePlusD, u, acc);   // x + y (incl. D*u)
    }
}

// ---------------------------------------------------------------------------
// Kernel 2: LayerNorm over last dim (512).  One wave32 per row, 16 elems per
// lane, shfl_xor reduction.  Optional residual input and optional f16 copy.
// ---------------------------------------------------------------------------
__global__ void ln_kernel(const float* __restrict__ in,
                          const float* __restrict__ res,    // may be null
                          const float* __restrict__ g,
                          const float* __restrict__ be,
                          float* __restrict__ out32,        // may be null
                          _Float16* __restrict__ out16) {   // may be null
    const int wave = threadIdx.x >> 5;
    const int lane = threadIdx.x & 31;
    const int row  = blockIdx.x * 8 + wave;   // 8 waves per block
    const size_t base = (size_t)row * DM;

    float v[16];
    float s = 0.0f;
#pragma unroll
    for (int i = 0; i < 16; ++i) {
        const int c = lane + 32 * i;
        float val = in[base + c];
        if (res) val += res[base + c];
        v[i] = val;
        s += val;
    }
#pragma unroll
    for (int off = 16; off > 0; off >>= 1) s += __shfl_xor(s, off, 32);
    const float mu = s * (1.0f / DM);

    float s2 = 0.0f;
#pragma unroll
    for (int i = 0; i < 16; ++i) { const float d = v[i] - mu; s2 += d * d; }
#pragma unroll
    for (int off = 16; off > 0; off >>= 1) s2 += __shfl_xor(s2, off, 32);
    const float rstd = rsqrtf(s2 * (1.0f / DM) + 1e-5f);

#pragma unroll
    for (int i = 0; i < 16; ++i) {
        const int c = lane + 32 * i;
        const float o = (v[i] - mu) * rstd * g[c] + be[c];
        if (out32) out32[base + c] = o;
        if (out16) out16[base + c] = (_Float16)o;
    }
}

// ---------------------------------------------------------------------------
// Kernel 3: f32 -> f16 weight conversion.
// ---------------------------------------------------------------------------
__global__ void cvt_f16_kernel(const float* __restrict__ in,
                               _Float16* __restrict__ out, int n) {
    const int i = blockIdx.x * blockDim.x + threadIdx.x;
    if (i < n) out[i] = (_Float16)in[i];
}

// ---------------------------------------------------------------------------
// Kernel 4: WMMA GEMM.  C[M,N] = act(A[M,K] * Bw[N,K]^T + bias).
// Block tile 128x128, 8 waves (wave tile 32x64 = 2x4 wmma tiles), K-step 32,
// v_wmma_f32_16x16x32_f16.  Double-buffered LDS staged by CDNA5 async paths:
//   A tile: per-lane global_load_async_to_lds_b128  (ASYNCcnt)
//   B tile: Tensor Data Mover tensor_load_to_lds    (TENSORcnt, wave 0)
// LDS layout (dynamic, base 0): A[2][128][32]f16 @ 0, B[2][128][32]f16 @ 16K.
// ---------------------------------------------------------------------------
#define GEMM_LDS_BYTES 32768
#define ABUF_BYTES 8192
#define BBUF_BASE 16384

template <bool RELU, typename OutT>
__global__ __launch_bounds__(256)
void gemm_kernel(const _Float16* __restrict__ A,
                 const _Float16* __restrict__ Bw,
                 const float* __restrict__ bias,
                 OutT* __restrict__ C,
                 int M, int N, int K) {
    extern __shared__ char smem[];
    const _Float16* lds16 = (const _Float16*)smem;

    const int tid  = threadIdx.x;
    const int lane = tid & 31;
    const int wave = tid >> 5;
    const int wm   = wave & 3;    // 4 waves along M
    const int wn   = wave >> 2;   // 2 waves along N
    const int m0   = blockIdx.y * 128;
    const int n0   = blockIdx.x * 128;

    v8f acc[2][4] = {};

    // global->LDS staging mapping: each thread brings 32B of the A tile
    const int lr = tid >> 1;            // tile row 0..127
    const int lc = (tid & 1) * 16;      // half offset 0 or 16

    const int mrow = lane & 15;
    const int kb   = (lane >> 4) * 8;

    const int nk = K >> 5;

    auto stage = [&](int buf, int k0) {
        // A tile: 128x32 halves via per-lane async b128 copies
        const _Float16* ga = A + (size_t)(m0 + lr) * K + k0 + lc;
        const unsigned aoff = (unsigned)(buf * ABUF_BYTES + (lr * 32 + lc) * 2);
        async_b128(aoff,      ga);
        async_b128(aoff + 16, ga + 8);
        // B tile: 128x32 halves via TDM (one descriptor, wave 0 only)
#if HAVE_TDM
        if (wave == 0) {
            tdm_load_tile_2d((unsigned)(BBUF_BASE + buf * ABUF_BYTES),
                             Bw + (size_t)n0 * K + k0,
                             /*tile_w=*/32, /*tile_h=*/128,
                             /*stride=*/(unsigned long long)K,
                             /*td0=*/(unsigned long long)K,
                             /*td1=*/128ull);
        }
#else
        const _Float16* gb = Bw + (size_t)(n0 + lr) * K + k0 + lc;
        const unsigned boff = (unsigned)(BBUF_BASE + buf * ABUF_BYTES + (lr * 32 + lc) * 2);
        async_b128(boff,      gb);
        async_b128(boff + 16, gb + 8);
#endif
    };

    auto wait_stage = [&]() {
        wait_asynccnt0();
#if HAVE_TDM
        if (wave == 0) __builtin_amdgcn_s_wait_tensorcnt(0);
#endif
        __syncthreads();
    };

    stage(0, 0);
    wait_stage();

    for (int kt = 0; kt < nk; ++kt) {
        const int cur = kt & 1;
        if (kt + 1 < nk) stage(cur ^ 1, (kt + 1) << 5);   // overlaps with WMMAs below

        const _Float16* As = lds16 + cur * (ABUF_BYTES / 2);
        const _Float16* Bs = lds16 + (BBUF_BASE / 2) + cur * (ABUF_BYTES / 2);

        F16x16 afrag[2], bfrag[4];
#pragma unroll
        for (int tm = 0; tm < 2; ++tm) {
            const int m = wm * 32 + tm * 16 + mrow;
            afrag[tm].h[0] = *(const v8h*)(As + m * 32 + kb);
            afrag[tm].h[1] = *(const v8h*)(As + m * 32 + kb + 16);
        }
#pragma unroll
        for (int tn = 0; tn < 4; ++tn) {
            const int n = wn * 64 + tn * 16 + mrow;
            bfrag[tn].h[0] = *(const v8h*)(Bs + n * 32 + kb);
            bfrag[tn].h[1] = *(const v8h*)(Bs + n * 32 + kb + 16);
        }
#pragma unroll
        for (int tm = 0; tm < 2; ++tm)
#pragma unroll
            for (int tn = 0; tn < 4; ++tn)
                acc[tm][tn] = __builtin_amdgcn_wmma_f32_16x16x32_f16(
                    false, afrag[tm].v, false, bfrag[tn].v,
                    (short)0, acc[tm][tn], false, false);

        wait_stage();   // next-stage copies done; safe to swap buffers
    }

    // Epilogue: C/D layout — lanes 0-15: M = r, lanes 16-31: M = 8 + r,
    // N = lane & 15 (per 16x16 f32 accumulator tile).
#pragma unroll
    for (int tm = 0; tm < 2; ++tm) {
#pragma unroll
        for (int tn = 0; tn < 4; ++tn) {
            const int col = n0 + wn * 64 + tn * 16 + (lane & 15);
            const float bv = bias[col];
#pragma unroll
            for (int r = 0; r < 8; ++r) {
                const int row = m0 + wm * 32 + tm * 16 + (lane >> 4) * 8 + r;
                float val = acc[tm][tn][r] + bv;
                if (RELU) val = fmaxf(val, 0.0f);
                C[(size_t)row * N + col] = (OutT)val;
            }
        }
    }
}

// ---------------------------------------------------------------------------
// Launch
// ---------------------------------------------------------------------------
extern "C" void kernel_launch(void* const* d_in, const int* in_sizes, int n_in,
                              void* d_out, int out_size, void* d_ws, size_t ws_size,
                              hipStream_t stream) {
    (void)in_sizes; (void)n_in; (void)out_size; (void)ws_size;

    const float* x      = (const float*)d_in[0];
    const float* log_dt = (const float*)d_in[1];
    const float* A_re   = (const float*)d_in[2];
    const float* A_im   = (const float*)d_in[3];
    const float* C_re   = (const float*)d_in[4];
    const float* C_im   = (const float*)d_in[5];
    const float* Dp     = (const float*)d_in[6];
    const float* g1     = (const float*)d_in[7];
    const float* be1    = (const float*)d_in[8];
    const float* g2     = (const float*)d_in[9];
    const float* be2    = (const float*)d_in[10];
    const float* W1     = (const float*)d_in[11];
    const float* b1     = (const float*)d_in[12];
    const float* W2     = (const float*)d_in[13];
    const float* b2     = (const float*)d_in[14];
    float* out          = (float*)d_out;

    // Workspace layout
    char* ws = (char*)d_ws;
    float*    xres = (float*)ws;                 ws += (size_t)ROWS * DM * 4;   // 16 MB
    float*    xn32 = (float*)ws;                 ws += (size_t)ROWS * DM * 4;   // 16 MB
    _Float16* xn16 = (_Float16*)ws;              ws += (size_t)ROWS * DM * 2;   //  8 MB
    _Float16* W1h  = (_Float16*)ws;              ws += (size_t)DFF * DM * 2;    //  2 MB
    _Float16* W2h  = (_Float16*)ws;              ws += (size_t)DM * DFF * 2;    //  2 MB
    _Float16* h1   = (_Float16*)ws;              ws += (size_t)ROWS * DFF * 2;  // 32 MB
    float*    y2   = (float*)ws;                 ws += (size_t)ROWS * DM * 4;   // 16 MB

    // 1) S4D recurrence + residual
    s4d_kernel<<<(BB * DM) / 256, 256, 0, stream>>>(x, log_dt, A_re, A_im,
                                                    C_re, C_im, Dp, xres);

    // 2) weight conversion to f16 (two GEMM operands)
    cvt_f16_kernel<<<(DFF * DM) / 256, 256, 0, stream>>>(W1, W1h, DFF * DM);
    cvt_f16_kernel<<<(DM * DFF) / 256, 256, 0, stream>>>(W2, W2h, DM * DFF);

    // 3) LayerNorm1 -> f32 (for final residual) + f16 (GEMM1 activation)
    ln_kernel<<<ROWS / 8, 256, 0, stream>>>(xres, nullptr, g1, be1, xn32, xn16);

    // 4) GEMM1: h1 = relu(xn16 * W1^T + b1)   [8192 x 2048]
    gemm_kernel<true, _Float16>
        <<<dim3(DFF / 128, ROWS / 128), 256, GEMM_LDS_BYTES, stream>>>(
            xn16, W1h, b1, h1, ROWS, DFF, DM);

    // 5) GEMM2: y2 = h1 * W2^T + b2           [8192 x 512]
    gemm_kernel<false, float>
        <<<dim3(DM / 128, ROWS / 128), 256, GEMM_LDS_BYTES, stream>>>(
            h1, W2h, b2, y2, ROWS, DM, DFF);

    // 6) LayerNorm2(xn32 + y2) -> out
    ln_kernel<<<ROWS / 8, 256, 0, stream>>>(xn32, y2, g2, be2, out, nullptr);
}
